// Model_39676907884213
// MI455X (gfx1250) — matrix operand order repaired and checked by reading, not compiled, not run on hardware
//
#include <hip/hip_runtime.h>

#define BM 128
#define BN 128
#define BK 32
#define LDT 40   // padded LDS row stride in bf16 elements (80 B, 16B-aligned)

typedef __attribute__((ext_vector_type(16))) __bf16 v16bf;
typedef __attribute__((ext_vector_type(8)))  __bf16 v8bf;
typedef __attribute__((ext_vector_type(8)))  float  v8f;
typedef unsigned short u16;
typedef unsigned int   u32;
typedef unsigned long long u64;

#if __has_builtin(__builtin_amdgcn_global_load_async_to_lds_b128) && \
    __has_builtin(__builtin_amdgcn_s_wait_asynccnt)
#define USE_ASYNC 1
#else
#define USE_ASYNC 0
#endif

// builtin expects: (int4 AS1*, int4 AS3*, imm offset, imm cpol) per round-2 diag
typedef int v4i_g __attribute__((vector_size(16)));
#define GLB_V4P(p) ((__attribute__((address_space(1))) v4i_g*)(u64)(p))
#define LDS_V4P(p) ((__attribute__((address_space(3))) v4i_g*)(u32)(u64)(p))

static __device__ __forceinline__ u16 f2bf(float f) {
  union { float f; u32 u; } v; v.f = f;
  u32 r = v.u + 0x7FFFu + ((v.u >> 16) & 1u);   // round-to-nearest-even
  return (u16)(r >> 16);
}

// -------- elementwise f32 -> bf16 (n must be a multiple of 2048) --------
__global__ __launch_bounds__(256)
void cvt_f32_bf16(const float* __restrict__ src, u16* __restrict__ dst, long long n)
{
  const long long i = ((long long)blockIdx.x * 256 + threadIdx.x) * 8;
  if (i >= n) return;
  const float4 a = *(const float4*)(src + i);
  const float4 b = *(const float4*)(src + i + 4);
  union { u16 s[8]; uint4 v; } o;
  o.s[0] = f2bf(a.x); o.s[1] = f2bf(a.y); o.s[2] = f2bf(a.z); o.s[3] = f2bf(a.w);
  o.s[4] = f2bf(b.x); o.s[5] = f2bf(b.y); o.s[6] = f2bf(b.z); o.s[7] = f2bf(b.w);
  *(uint4*)(dst + i) = o.v;
}

// Generic batched GEMM:  O[m,n] = sum_k A[m,k]*Bn[n,k] (+bias[n]) (/divm[m,n])
// A,Bn bf16; A row-major [M,K], Bn N-major [N,K]. Output f32 or bf16,
// row-major (transO=0) or transposed (transO=1). bf16 WMMA, f32 accumulate.
__global__ __launch_bounds__(256)
void gemm_bf16_wmma(const u16* __restrict__ A, long long sA, int lda,
                    const u16* __restrict__ Bn, long long sB, int ldb,
                    const float* __restrict__ bias,
                    const float* __restrict__ divm, long long sD, int ldd,
                    void* __restrict__ O, long long sO, int ldo,
                    int transO, int outBf16, int K)
{
  __shared__ __align__(16) u16 As[2][BM][LDT];
  __shared__ __align__(16) u16 Bs[2][BN][LDT];

  const int tid   = threadIdx.x;
  const int lane  = tid & 31;
  const int wid   = tid >> 5;     // 0..7
  const int waveM = wid >> 1;     // 0..3 -> 32 rows
  const int waveN = wid & 1;      // 0..1 -> 64 cols
  const int hlf   = lane >> 4;    // lane half
  const int l15   = lane & 15;

  const int bm = blockIdx.y * BM;
  const int bn = blockIdx.x * BN;
  const long long z = blockIdx.z;

  const u16* Ab = A  + z * sA;
  const u16* Bb = Bn + z * sB;

  v8f acc[2][4];
#pragma unroll
  for (int i = 0; i < 2; ++i)
#pragma unroll
    for (int j = 0; j < 4; ++j)
      acc[i][j] = (v8f){0.f,0.f,0.f,0.f,0.f,0.f,0.f,0.f};

  // staging map: 256 thr -> 64 rows x 4 chunks of 8 bf16, two passes (128 rows)
  const int srow = tid >> 2;        // 0..63
  const int scol = (tid & 3) * 8;   // 0,8,16,24

  const int nK = K / BK;

#if USE_ASYNC
  auto stage = [&](int kt, int b) {
#pragma unroll
    for (int p = 0; p < 2; ++p) {
      const int row = srow + 64 * p;
      __builtin_amdgcn_global_load_async_to_lds_b128(
          GLB_V4P(Ab + (long long)(bm + row) * lda + kt * BK + scol),
          LDS_V4P(&As[b][row][scol]), 0, 0);
      __builtin_amdgcn_global_load_async_to_lds_b128(
          GLB_V4P(Bb + (long long)(bn + row) * ldb + kt * BK + scol),
          LDS_V4P(&Bs[b][row][scol]), 0, 0);
    }
  };
  stage(0, 0);
#else
  uint4 ra[2], rb[2];
  auto loadRegs = [&](int kt) {
#pragma unroll
    for (int p = 0; p < 2; ++p) {
      const int row = srow + 64 * p;
      ra[p] = *(const uint4*)(Ab + (long long)(bm + row) * lda + kt * BK + scol);
      rb[p] = *(const uint4*)(Bb + (long long)(bn + row) * ldb + kt * BK + scol);
    }
  };
  loadRegs(0);
#endif

  for (int kt = 0; kt < nK; ++kt) {
    const int cb = kt & 1;
#if USE_ASYNC
    __builtin_amdgcn_s_wait_asynccnt(0);   // my tile-kt writes to LDS landed
    __syncthreads();                       // everyone's landed; buf cb^1 free
    if (kt + 1 < nK) stage(kt + 1, cb ^ 1);
#else
#pragma unroll
    for (int p = 0; p < 2; ++p) {
      const int row = srow + 64 * p;
      *(uint4*)&As[cb][row][scol] = ra[p];
      *(uint4*)&Bs[cb][row][scol] = rb[p];
    }
    __syncthreads();
    if (kt + 1 < nK) loadRegs(kt + 1);     // overlap next-tile loads w/ compute
#endif

    // --- fragments per ISA wave32 VGPR layouts ---
    // A 16x32: lanes 0-15 K={0..7,16..23}, lanes 16-31 K={8..15,24..31}
    const int alo = hlf * 8;
    v16bf afrag[2];
#pragma unroll
    for (int mi = 0; mi < 2; ++mi) {
      const int arow = waveM * 32 + mi * 16 + l15;
      const v8bf a0 = *reinterpret_cast<const v8bf*>(&As[cb][arow][alo]);
      const v8bf a1 = *reinterpret_cast<const v8bf*>(&As[cb][arow][16 + alo]);
      afrag[mi] = __builtin_shufflevector(a0, a1, 0,1,2,3,4,5,6,7,8,9,10,11,12,13,14,15);
    }
    // B 32x16: lanes 0-15 K={0..7,16..23}, lanes 16-31 K={8..15,24..31}, N = lane&15
    const int bko = hlf * 8;
    v16bf bfrag[4];
#pragma unroll
    for (int ni = 0; ni < 4; ++ni) {
      const int brow = waveN * 64 + ni * 16 + l15;
      const v8bf b0 = *reinterpret_cast<const v8bf*>(&Bs[cb][brow][bko]);
      const v8bf b1 = *reinterpret_cast<const v8bf*>(&Bs[cb][brow][bko + 16]);
      bfrag[ni] = __builtin_shufflevector(b0, b1, 0,1,2,3,4,5,6,7,8,9,10,11,12,13,14,15);
    }

#pragma unroll
    for (int mi = 0; mi < 2; ++mi)
#pragma unroll
      for (int ni = 0; ni < 4; ++ni)
        acc[mi][ni] = __builtin_amdgcn_wmma_f32_16x16x32_bf16(
            false, afrag[mi], false, bfrag[ni],
            (short)0, acc[mi][ni], false, false);
  }

  // --- epilogue: C/D layout — VGPR rr: M = rr + 8*hlf, N = lane&15 ---
#pragma unroll
  for (int mi = 0; mi < 2; ++mi)
#pragma unroll
    for (int ni = 0; ni < 4; ++ni) {
      const v8f a = acc[mi][ni];
      const int nIdx = bn + waveN * 64 + ni * 16 + l15;
      const float bv = bias ? bias[nIdx] : 0.f;
#pragma unroll
      for (int rr = 0; rr < 8; ++rr) {
        const int mIdx = bm + waveM * 32 + mi * 16 + rr + 8 * hlf;
        float val = a[rr] + bv;
        if (divm) val /= divm[z * sD + (long long)mIdx * ldd + nIdx];
        const long long off = z * sO +
            (transO ? ((long long)nIdx * ldo + mIdx)
                    : ((long long)mIdx * ldo + nIdx));
        if (outBf16) ((u16*)O)[off] = f2bf(val);
        else        ((float*)O)[off] = val;
      }
    }
}

// row softmax: reads f32 qk rows (cols==1024), writes bf16 attn
__global__ __launch_bounds__(256)
void softmax_rows(const float* __restrict__ qk, u16* __restrict__ attn, int cols)
{
  const long long row = blockIdx.x;
  const float* p = qk + row * (long long)cols;
  const int tid = threadIdx.x;
  float4 v = reinterpret_cast<const float4*>(p)[tid];
  __shared__ float red[256];

  red[tid] = fmaxf(fmaxf(v.x, v.y), fmaxf(v.z, v.w));
  __syncthreads();
  for (int s = 128; s > 0; s >>= 1) {
    if (tid < s) red[tid] = fmaxf(red[tid], red[tid + s]);
    __syncthreads();
  }
  const float rmax = red[0];
  __syncthreads();

  v.x = __expf(v.x - rmax);
  v.y = __expf(v.y - rmax);
  v.z = __expf(v.z - rmax);
  v.w = __expf(v.w - rmax);
  red[tid] = v.x + v.y + v.z + v.w;
  __syncthreads();
  for (int s = 128; s > 0; s >>= 1) {
    if (tid < s) red[tid] += red[tid + s];
    __syncthreads();
  }
  const float rinv = 1.0f / red[0];
  ushort4 w;
  w.x = f2bf(v.x * rinv); w.y = f2bf(v.y * rinv);
  w.z = f2bf(v.z * rinv); w.w = f2bf(v.w * rinv);
  reinterpret_cast<ushort4*>(attn + row * (long long)cols)[tid] = w;
}

extern "C" void kernel_launch(void* const* d_in, const int* in_sizes, int n_in,
                              void* d_out, int out_size, void* d_ws, size_t ws_size,
                              hipStream_t stream)
{
  (void)in_sizes; (void)n_in; (void)out_size; (void)ws_size;
  const int Bv = 4, SQ = 2048, SK = 1024, D = 1024;

  const float* x  = (const float*)d_in[0];
  const float* y  = (const float*)d_in[1];
  // d_in[2]/[3] = W1/b1: dead code in reference
  const float* W2 = (const float*)d_in[4];
  const float* b2 = (const float*)d_in[5];
  const float* W3 = (const float*)d_in[6];
  const float* b3 = (const float*)d_in[7];
  const float* W4 = (const float*)d_in[8];
  const float* b4 = (const float*)d_in[9];
  const float* W5 = (const float*)d_in[10];
  const float* b5 = (const float*)d_in[11];
  float* out = (float*)d_out;

  const long long nKV = (long long)Bv * SK * D;   // 4M
  const long long nQ  = (long long)Bv * SQ * D;   // 8M
  const long long nW  = (long long)D * D;         // 1M

  // f32 region first (alignment), then bf16 region
  float* inv = (float*)d_ws;           // [B,SQ,D]  f32
  float* qk  = inv + nQ;               // [B,SQ,SK] f32
  u16* xbf   = (u16*)(qk + nQ);        // [B,SQ,D]
  u16* ybf   = xbf + nQ;               // [B,SK,D]
  u16* w2b   = ybf + nKV;
  u16* w3b   = w2b + nW;
  u16* w4b   = w3b + nW;
  u16* w5b   = w4b + nW;
  u16* kbf   = w5b + nW;               // [B,SK,D]
  u16* vTb   = kbf + nKV;              // [B,D,SK]  (transposed v)
  u16* qyb   = vTb + nKV;              // [B,SQ,D]
  u16* attnb = qyb + nQ;               // [B,SQ,SK]

  const dim3 blk(256);

  // ---- pre-pass: f32 -> bf16 ----
  cvt_f32_bf16<<<dim3((unsigned)(nQ  / 2048)), blk, 0, stream>>>(x,  xbf, nQ);
  cvt_f32_bf16<<<dim3((unsigned)(nKV / 2048)), blk, 0, stream>>>(y,  ybf, nKV);
  cvt_f32_bf16<<<dim3((unsigned)(nW  / 2048)), blk, 0, stream>>>(W2, w2b, nW);
  cvt_f32_bf16<<<dim3((unsigned)(nW  / 2048)), blk, 0, stream>>>(W3, w3b, nW);
  cvt_f32_bf16<<<dim3((unsigned)(nW  / 2048)), blk, 0, stream>>>(W4, w4b, nW);
  cvt_f32_bf16<<<dim3((unsigned)(nW  / 2048)), blk, 0, stream>>>(W5, w5b, nW);

  // k = y @ W2^T + b2  -> bf16            (W [e,d] is N-major)
  gemm_bf16_wmma<<<dim3(D/BN, SK/BM, Bv), blk, 0, stream>>>(
      ybf, (long long)SK*D, D,  w2b, 0, D,  b2,  nullptr, 0, 0,
      kbf, (long long)SK*D, D, 0, 1,  D);

  // vT = (y @ W3^T + b3)^T -> bf16 [B,D,SK]
  gemm_bf16_wmma<<<dim3(D/BN, SK/BM, Bv), blk, 0, stream>>>(
      ybf, (long long)SK*D, D,  w3b, 0, D,  b3,  nullptr, 0, 0,
      vTb, (long long)D*SK, SK, 1, 1,  D);

  // qy = x @ W4^T + b4 -> bf16
  gemm_bf16_wmma<<<dim3(D/BN, SQ/BM, Bv), blk, 0, stream>>>(
      xbf, (long long)SQ*D, D,  w4b, 0, D,  b4,  nullptr, 0, 0,
      qyb, (long long)SQ*D, D, 0, 1,  D);

  // inv_scale = x @ W5^T + b5 -> f32
  gemm_bf16_wmma<<<dim3(D/BN, SQ/BM, Bv), blk, 0, stream>>>(
      xbf, (long long)SQ*D, D,  w5b, 0, D,  b5,  nullptr, 0, 0,
      inv, (long long)SQ*D, D, 0, 0,  D);

  // qk = (qy @ k^T) / inv_scale -> f32    (k[s,d] is N-major: n=s,k=d)
  gemm_bf16_wmma<<<dim3(SK/BN, SQ/BM, Bv), blk, 0, stream>>>(
      qyb, (long long)SQ*D, D,  kbf, (long long)SK*D, D,  nullptr,
      inv, (long long)SQ*D, D,
      qk, (long long)SQ*SK, SK, 0, 0,  D);

  // attn = softmax(qk) -> bf16
  softmax_rows<<<dim3(Bv*SQ), blk, 0, stream>>>(qk, attnb, SK);

  // out = attn @ v -> f32                 (vT[d,s] is N-major: n=d,k=s)
  gemm_bf16_wmma<<<dim3(D/BN, SQ/BM, Bv), blk, 0, stream>>>(
      attnb, (long long)SQ*SK, SK,  vTb, (long long)D*SK, SK,  nullptr, nullptr, 0, 0,
      out, (long long)SQ*D, D, 0, 0,  SK);
}
